// EndPredictor_395136991244
// MI455X (gfx1250) — compile-verified, matrix operand order checked by reading, not run on hardware
//
#include <hip/hip_runtime.h>
#include <stdint.h>

// Problem sizes (fixed by the reference)
#define BB 256
#define TT 512
#define DD 1024
#define UU 1024
#define N3 3072
#define N2 2048
#define LN_EPS 1e-5f

typedef __bf16 bf16x8  __attribute__((ext_vector_type(8)));
typedef __bf16 v16bf   __attribute__((ext_vector_type(16)));
typedef float  v8f     __attribute__((ext_vector_type(8)));
typedef uint32_t u32x4 __attribute__((ext_vector_type(4)));
typedef uint32_t u32x8 __attribute__((ext_vector_type(8)));

// ---------- helpers ----------
__device__ __forceinline__ uint16_t f2bf(float f) {
    uint32_t u = __builtin_bit_cast(uint32_t, f);
    u = (u + 0x7FFFu + ((u >> 16) & 1u)) >> 16;   // round-to-nearest-even
    return (uint16_t)u;
}
__device__ __forceinline__ float bf2f(uint16_t h) {
    uint32_t u = ((uint32_t)h) << 16;
    return __builtin_bit_cast(float, u);
}

// Load a 16x32 bf16 A-fragment (or, identically, a B-fragment from an N-major
// matrix) per CDNA5 ISA §7.12.2: lanes 0-15 hold K=[0..7]∪[16..23] of row (lane&15),
// lanes 16-31 hold K=[8..15]∪[24..31].  Two 16-byte contiguous loads per lane.
// Works for global (global_load_b128) and LDS (ds_read_b128) source pointers.
__device__ __forceinline__ v16bf load_frag(const uint16_t* __restrict__ base,
                                           int ld, int row, int k0, int lane) {
    const uint16_t* p = base + (size_t)row * ld + k0 + ((lane & 16) ? 8 : 0);
    union { v16bf v; bf16x8 h[2]; } u;
    u.h[0] = *reinterpret_cast<const bf16x8*>(p);
    u.h[1] = *reinterpret_cast<const bf16x8*>(p + 16);
    return u.v;
}

// Issue a 2-D TDM copy (global -> LDS) per ISA §8 (D# group0 + group1 only:
// 2D tensor, groups 2/3 disabled via the 2-operand tensor_load_to_lds form).
// Copies `rows` x 32 bf16 elements; global rows are `row_stride` elems apart;
// LDS destination is contiguous rows of 64B.  Wave-level; EXEC-independent.
__device__ __forceinline__ void tdm_load_2d(uint32_t lds_addr, const void* gptr,
                                            uint32_t rows, uint32_t row_stride) {
    uint64_t ga = (uint64_t)(uintptr_t)gptr;
    uint32_t ga_lo = __builtin_amdgcn_readfirstlane((uint32_t)ga);
    uint32_t ga_hi = __builtin_amdgcn_readfirstlane((uint32_t)(ga >> 32));
    u32x4 g0;
    g0[0] = 1u;                                   // count=1, user descriptor
    g0[1] = lds_addr;                             // lds_addr  (bits 63:32)
    g0[2] = ga_lo;                                // global_addr[31:0]
    g0[3] = (ga_hi & 0x01FFFFFFu) | (2u << 30);   // global_addr[56:32] | type=2
    u32x8 g1;
    g1[0] = 0x00010000u;                          // workgroup_mask=0, data_size=1 (2B)
    g1[1] = (1024u & 0xFFFFu) << 16;              // tensor_dim0[15:0] @bits63:48
    g1[2] = (3072u & 0xFFFFu) << 16;              // tensor_dim1[15:0] @bits95:80
    g1[3] = 32u << 16;                            // tile_dim0=32      @bits127:112
    g1[4] = rows & 0xFFFFu;                       // tile_dim1         @bits143:128
    g1[5] = row_stride;                           // tensor_dim0_stride[31:0]
    g1[6] = 0u;                                   // stride hi / dim1_stride lo
    g1[7] = 0u;
    asm volatile("tensor_load_to_lds %0, %1" :: "s"(g0), "s"(g1) : "memory");
}

// ---------- one-time precision/layout conversion ----------
__global__ void k_cvt(const float* __restrict__ in, uint16_t* __restrict__ out, int n) {
    int i = (blockIdx.x * 256 + threadIdx.x) * 4;
    if (i + 3 < n) {
        float4 v = *reinterpret_cast<const float4*>(in + i);
        ushort4 o;
        o.x = f2bf(v.x); o.y = f2bf(v.y); o.z = f2bf(v.z); o.w = f2bf(v.w);
        *reinterpret_cast<ushort4*>(out + i) = o;
    }
}

// Transpose (K x N) fp32 -> (N x K) bf16 so B-fragments load row-contiguously.
__global__ void k_cvtT(const float* __restrict__ in, uint16_t* __restrict__ out,
                       int K, int N) {
    int o = blockIdx.x * 256 + threadIdx.x;       // o = n*K + k
    int n = o / K;
    int k = o - n * K;
    out[o] = f2bf(in[(size_t)k * N + n]);
}

// ---------- phase 1: s1 = LN(x @ W + b), stored bf16 as (T, B, 3U) ----------
// One block = 16 M-rows x full N=3072.  16 waves, 12 N-tiles per wave.
__global__ __launch_bounds__(512, 1)
void k_s1(const uint16_t* __restrict__ xb, const uint16_t* __restrict__ Wt,
          const float* __restrict__ bias, const float* __restrict__ gamma0,
          const float* __restrict__ beta0, uint16_t* __restrict__ s1) {
    __shared__ float ssum[16], ssq[16];
    const int tid  = threadIdx.x;
    const int lane = tid & 31;
    const int w    = tid >> 5;
    const int m0   = blockIdx.x * 16;
    const int l15  = lane & 15;

    v8f acc[12];
#pragma unroll
    for (int j = 0; j < 12; ++j)
#pragma unroll
        for (int e = 0; e < 8; ++e) acc[j][e] = 0.0f;

    const int arow = m0 + l15;
    for (int kc = 0; kc < DD; kc += 32) {
        v16bf a = load_frag(xb, DD, arow, kc, lane);
        if (kc + 32 < DD)
            __builtin_prefetch(xb + (size_t)arow * DD + kc + 32, 0, 1);
        // software-pipelined B fragments: issue tile j+1 before consuming j
        v16bf bcur = load_frag(Wt, DD, (w * 12) * 16 + l15, kc, lane);
#pragma unroll
        for (int j = 0; j < 12; ++j) {
            v16bf bnext = bcur;
            if (j < 11)
                bnext = load_frag(Wt, DD, (w * 12 + j + 1) * 16 + l15, kc, lane);
            acc[j] = __builtin_amdgcn_wmma_f32_16x16x32_bf16(
                false, a, false, bcur, (short)0, acc[j], false, false);
            bcur = bnext;
        }
    }

    if (tid < 16) { ssum[tid] = 0.0f; ssq[tid] = 0.0f; }
    __syncthreads();

    float psum[8], psq[8];
#pragma unroll
    for (int e = 0; e < 8; ++e) { psum[e] = 0.0f; psq[e] = 0.0f; }
#pragma unroll
    for (int j = 0; j < 12; ++j) {
        const int n = (w * 12 + j) * 16 + l15;
        const float bv = bias[n];
#pragma unroll
        for (int e = 0; e < 8; ++e) {
            float v = acc[j][e] + bv;
            acc[j][e] = v;
            psum[e] += v;
            psq[e]  += v * v;
        }
    }
    const int mh = (lane >> 4) * 8;
#pragma unroll
    for (int e = 0; e < 8; ++e) {
        atomicAdd(&ssum[mh + e], psum[e]);   // ds_add_f32
        atomicAdd(&ssq[mh + e],  psq[e]);
    }
    __syncthreads();

    const float invN = 1.0f / (float)N3;
#pragma unroll
    for (int e = 0; e < 8; ++e) {
        const int mloc = mh + e;
        float mean = ssum[mloc] * invN;
        float var  = ssq[mloc] * invN - mean * mean;
        var = var < 0.0f ? 0.0f : var;
        const float inv = 1.0f / (sqrtf(var + LN_EPS) + LN_EPS);
        const int r = m0 + mloc;            // r = b*T + t (x is (B,T,D) row-major)
        const int bidx = r >> 9;            // /512
        const int t    = r & 511;
        const size_t obase = ((size_t)t * BB + bidx) * N3;
#pragma unroll
        for (int j = 0; j < 12; ++j) {
            const int n = (w * 12 + j) * 16 + l15;
            float v = gamma0[n] * (acc[j][e] - mean) * inv + beta0[n];
            s1[obase + n] = f2bf(v);
        }
    }
}

// ---------- phase 2a: s2 = LN(h @ U[:, :2U]); s = hsig(s1_t + s2) -> z, r*h ----------
// One block = 16 batch rows x full N=2048.  16 waves, 8 N-tiles (=128 rows of U^T)
// per wave.  Each wave TDM-stages its private 128x32 bf16 B sub-panel into LDS,
// double-buffered across the K loop; no cross-wave sharing -> no barriers in loop.
__global__ __launch_bounds__(512, 1)
void k_zr(const uint16_t* __restrict__ hb, const uint16_t* __restrict__ Ut,
          const uint16_t* __restrict__ s1, const float* __restrict__ h32,
          const float* __restrict__ gam1, const float* __restrict__ bet1,
          float* __restrict__ zbuf, uint16_t* __restrict__ rhb, int t) {
    __shared__ uint16_t bpanel[2][16][128 * 32];   // 256 KB (<= 320 KB/WGP)
    __shared__ float ssum[16], ssq[16];
    const int tid  = threadIdx.x;
    const int lane = tid & 31;
    const int w    = tid >> 5;
    const int ws   = __builtin_amdgcn_readfirstlane(w);   // uniform wave id (SGPR)
    const int m0   = blockIdx.x * 16;
    const int l15  = lane & 15;

    v8f acc[8];
#pragma unroll
    for (int j = 0; j < 8; ++j)
#pragma unroll
        for (int e = 0; e < 8; ++e) acc[j][e] = 0.0f;

    // This wave's 128 U^T rows start here (N-major, ld = 1024).
    const uint16_t* gpan = Ut + (size_t)(ws * 128) * UU;
    const uint32_t lds0 =
        __builtin_amdgcn_readfirstlane((uint32_t)(uintptr_t)&bpanel[0][ws][0]);
    const uint32_t lds1 =
        __builtin_amdgcn_readfirstlane((uint32_t)(uintptr_t)&bpanel[1][ws][0]);

    tdm_load_2d(lds0, gpan, 128u, (uint32_t)UU);           // prime buffer 0
    const int arow = m0 + l15;
    for (int kc = 0, it = 0; kc < UU; kc += 32, ++it) {
        const int cur = it & 1;
        if (kc + 32 < UU) {                                 // prefetch next chunk
            tdm_load_2d(cur ? lds0 : lds1, gpan + (kc + 32), 128u, (uint32_t)UU);
            __builtin_amdgcn_s_wait_tensorcnt(1);           // current chunk done
        } else {
            __builtin_amdgcn_s_wait_tensorcnt(0);           // last chunk done
        }
        asm volatile("" ::: "memory");                      // fence LDS reads below
        v16bf a = load_frag(hb, UU, arow, kc, lane);
        const uint16_t* bp = &bpanel[cur][ws][0];           // ld = 32, k0 = 0
#pragma unroll
        for (int j = 0; j < 8; ++j) {
            v16bf bf = load_frag(bp, 32, j * 16 + l15, 0, lane);   // ds_read_b128 x2
            acc[j] = __builtin_amdgcn_wmma_f32_16x16x32_bf16(
                false, a, false, bf, (short)0, acc[j], false, false);
        }
    }

    if (tid < 16) { ssum[tid] = 0.0f; ssq[tid] = 0.0f; }
    __syncthreads();

    float psum[8], psq[8];
#pragma unroll
    for (int e = 0; e < 8; ++e) { psum[e] = 0.0f; psq[e] = 0.0f; }
#pragma unroll
    for (int j = 0; j < 8; ++j)
#pragma unroll
        for (int e = 0; e < 8; ++e) {
            float v = acc[j][e];
            psum[e] += v;
            psq[e]  += v * v;
        }
    const int mh = (lane >> 4) * 8;
#pragma unroll
    for (int e = 0; e < 8; ++e) {
        atomicAdd(&ssum[mh + e], psum[e]);
        atomicAdd(&ssq[mh + e],  psq[e]);
    }
    __syncthreads();

    const float invN = 1.0f / (float)N2;
#pragma unroll
    for (int e = 0; e < 8; ++e) {
        const int mloc = mh + e;
        float mean = ssum[mloc] * invN;
        float var  = ssq[mloc] * invN - mean * mean;
        var = var < 0.0f ? 0.0f : var;
        const float inv = 1.0f / (sqrtf(var + LN_EPS) + LN_EPS);
        const int m = m0 + mloc;
        const size_t s1base = ((size_t)t * BB + m) * N3;
#pragma unroll
        for (int j = 0; j < 8; ++j) {
            const int n = (w * 8 + j) * 16 + l15;
            float s2  = gam1[n] * (acc[j][e] - mean) * inv + bet1[n];
            float s1v = bf2f(s1[s1base + n]);
            float s   = 0.2f * (s1v + s2) + 0.5f;
            s = s < 0.0f ? 0.0f : (s > 1.0f ? 1.0f : s);
            if (n < UU) {
                zbuf[(size_t)m * UU + n] = s;                       // z
            } else {
                const int nn = n - UU;                              // r
                float hv = h32[(size_t)m * UU + nn];
                rhb[(size_t)m * UU + nn] = f2bf(s * hv);            // r*h (bf16)
            }
        }
    }
}

// ---------- phase 2b: h = gate(z, tanh(s1_t + LN((r*h) @ U[:, 2U:]))) ----------
// One block = 16 batch rows x full N=1024.  8 waves, 8 N-tiles per wave.
// Same per-wave TDM double-buffered staging of the U^T[2U:] panel.
__global__ __launch_bounds__(256, 1)
void k_hh(const uint16_t* __restrict__ rhb, const uint16_t* __restrict__ Ut3,
          const uint16_t* __restrict__ s1, const float* __restrict__ gam,
          const float* __restrict__ bet, const float* __restrict__ zbuf,
          const uint8_t* __restrict__ mask, float* __restrict__ h32,
          uint16_t* __restrict__ hb, int t) {
    __shared__ uint16_t bpanel[2][8][128 * 32];    // 128 KB
    __shared__ float ssum[16], ssq[16];
    const int tid  = threadIdx.x;
    const int lane = tid & 31;
    const int w    = tid >> 5;
    const int ws   = __builtin_amdgcn_readfirstlane(w);
    const int m0   = blockIdx.x * 16;
    const int l15  = lane & 15;

    v8f acc[8];
#pragma unroll
    for (int j = 0; j < 8; ++j)
#pragma unroll
        for (int e = 0; e < 8; ++e) acc[j][e] = 0.0f;

    const uint16_t* gpan = Ut3 + (size_t)(ws * 128) * UU;
    const uint32_t lds0 =
        __builtin_amdgcn_readfirstlane((uint32_t)(uintptr_t)&bpanel[0][ws][0]);
    const uint32_t lds1 =
        __builtin_amdgcn_readfirstlane((uint32_t)(uintptr_t)&bpanel[1][ws][0]);

    tdm_load_2d(lds0, gpan, 128u, (uint32_t)UU);
    const int arow = m0 + l15;
    for (int kc = 0, it = 0; kc < UU; kc += 32, ++it) {
        const int cur = it & 1;
        if (kc + 32 < UU) {
            tdm_load_2d(cur ? lds0 : lds1, gpan + (kc + 32), 128u, (uint32_t)UU);
            __builtin_amdgcn_s_wait_tensorcnt(1);
        } else {
            __builtin_amdgcn_s_wait_tensorcnt(0);
        }
        asm volatile("" ::: "memory");
        v16bf a = load_frag(rhb, UU, arow, kc, lane);
        const uint16_t* bp = &bpanel[cur][ws][0];
#pragma unroll
        for (int j = 0; j < 8; ++j) {
            v16bf bf = load_frag(bp, 32, j * 16 + l15, 0, lane);
            acc[j] = __builtin_amdgcn_wmma_f32_16x16x32_bf16(
                false, a, false, bf, (short)0, acc[j], false, false);
        }
    }

    if (tid < 16) { ssum[tid] = 0.0f; ssq[tid] = 0.0f; }
    __syncthreads();

    float psum[8], psq[8];
#pragma unroll
    for (int e = 0; e < 8; ++e) { psum[e] = 0.0f; psq[e] = 0.0f; }
#pragma unroll
    for (int j = 0; j < 8; ++j)
#pragma unroll
        for (int e = 0; e < 8; ++e) {
            float v = acc[j][e];
            psum[e] += v;
            psq[e]  += v * v;
        }
    const int mh = (lane >> 4) * 8;
#pragma unroll
    for (int e = 0; e < 8; ++e) {
        atomicAdd(&ssum[mh + e], psum[e]);
        atomicAdd(&ssq[mh + e],  psq[e]);
    }
    __syncthreads();

    const float invN = 1.0f / (float)UU;
#pragma unroll
    for (int e = 0; e < 8; ++e) {
        const int mloc = mh + e;
        float mean = ssum[mloc] * invN;
        float var  = ssq[mloc] * invN - mean * mean;
        var = var < 0.0f ? 0.0f : var;
        const float inv = 1.0f / (sqrtf(var + LN_EPS) + LN_EPS);
        const int m = m0 + mloc;
        const bool mk = mask[(size_t)m * TT + t] != 0;
        const size_t s1base = ((size_t)t * BB + m) * N3 + N2;
#pragma unroll
        for (int j = 0; j < 8; ++j) {
            const int n = (w * 8 + j) * 16 + l15;
            float lnv  = gam[n] * (acc[j][e] - mean) * inv + bet[n];
            float s1v  = bf2f(s1[s1base + n]);
            float hc   = tanhf(s1v + lnv);
            float z    = zbuf[(size_t)m * UU + n];
            float hold = h32[(size_t)m * UU + n];
            float hnew = z * hold + (1.0f - z) * hc;
            float ho   = mk ? hnew : hold;
            h32[(size_t)m * UU + n] = ho;
            hb[(size_t)m * UU + n]  = f2bf(ho);
        }
    }
}

// ---------- final: sigmoid(h @ W1 + b1) ----------
__global__ void k_final(const float* __restrict__ h32, const float* __restrict__ W1,
                        const float* __restrict__ b1, float* __restrict__ out) {
    const int m = threadIdx.x;            // 256 threads, one per batch row
    float s = b1[0];
    for (int k = 0; k < UU; ++k) s += h32[(size_t)m * UU + k] * W1[k];
    out[m] = 1.0f / (1.0f + expf(-s));
}

// ---------- driver ----------
extern "C" void kernel_launch(void* const* d_in, const int* in_sizes, int n_in,
                              void* d_out, int out_size, void* d_ws, size_t ws_size,
                              hipStream_t stream) {
    const float*   x      = (const float*)d_in[0];
    const uint8_t* mask   = (const uint8_t*)d_in[1];
    const float*   W      = (const float*)d_in[2];
    const float*   U      = (const float*)d_in[3];
    const float*   bias   = (const float*)d_in[4];
    const float*   gammas = (const float*)d_in[5];
    const float*   betas  = (const float*)d_in[6];
    const float*   W1     = (const float*)d_in[7];
    const float*   b1     = (const float*)d_in[8];
    float*         out    = (float*)d_out;

    char* p = (char*)d_ws;
    auto alloc = [&](size_t bytes) -> char* {
        char* r = p;
        p += (bytes + 255) & ~(size_t)255;
        return r;
    };
    uint16_t* xb  = (uint16_t*)alloc((size_t)BB * TT * DD * 2);   // x in bf16 (M x K)
    uint16_t* Wt  = (uint16_t*)alloc((size_t)N3 * DD * 2);        // W^T bf16 (N x K)
    uint16_t* Ut  = (uint16_t*)alloc((size_t)N3 * UU * 2);        // U^T bf16 (N x K)
    uint16_t* s1  = (uint16_t*)alloc((size_t)TT * BB * N3 * 2);   // s1_all bf16 (T,B,3U)
    float*    h32 = (float*)   alloc((size_t)BB * UU * 4);        // h fp32
    uint16_t* hb  = (uint16_t*)alloc((size_t)BB * UU * 2);        // h bf16
    float*    zb  = (float*)   alloc((size_t)BB * UU * 4);        // z
    uint16_t* rhb = (uint16_t*)alloc((size_t)BB * UU * 2);        // r*h bf16

    const int nx = BB * TT * DD;
    k_cvt <<<nx / 1024, 256, 0, stream>>>(x, xb, nx);
    k_cvtT<<<(N3 * DD) / 256, 256, 0, stream>>>(W, Wt, DD, N3);
    k_cvtT<<<(N3 * UU) / 256, 256, 0, stream>>>(U, Ut, UU, N3);
    (void)hipMemsetAsync(h32, 0, (size_t)BB * UU * 4, stream);
    (void)hipMemsetAsync(hb,  0, (size_t)BB * UU * 2, stream);

    k_s1<<<(BB * TT) / 16, 512, 0, stream>>>(xb, Wt, bias, gammas, betas, s1);

    const uint16_t* Ut3 = Ut + (size_t)N2 * UU;
    for (int t = 0; t < TT; ++t) {
        k_zr<<<BB / 16, 512, 0, stream>>>(hb, Ut, s1, h32,
                                          gammas + N3, betas + N3, zb, rhb, t);
        k_hh<<<BB / 16, 256, 0, stream>>>(rhb, Ut3, s1,
                                          gammas + N3 + N2, betas + N3 + N2,
                                          zb, mask, h32, hb, t);
    }
    k_final<<<1, 256, 0, stream>>>(h32, W1, b1, out);
}
